// SelfAttention_68779606278609
// MI455X (gfx1250) — compile-verified, matrix operand order checked by reading, not run on hardware
//
#include <hip/hip_runtime.h>

typedef __bf16 bf16;
typedef __attribute__((ext_vector_type(16))) __bf16 v16bf;
typedef __attribute__((ext_vector_type(2)))  __bf16 bf16x2;
typedef __attribute__((ext_vector_type(8)))  float  v8f;

// ---------------------------------------------------------------------------
// Fragment helpers (CDNA5 WMMA 16x16x32 bf16 layouts, wave32)
// A (16x32, MxK): lane holds row = lane&15; lanes 0-15 hold K 0..7,16..23,
//                 lanes 16-31 hold K 8..15,24..31 (pairs per VGPR).
// B (32x16, KxN): lane holds col = lane&15; same K distribution.
// C (16x16 f32) : lane holds col = lane&15; VGPR v holds row v (+8 upper half).
// ---------------------------------------------------------------------------
__device__ __forceinline__ int frag_k(int lane, int p) {
  int koff = (lane >> 4) << 3;                       // 0 or 8
  return koff + ((p & 3) << 1) + ((p >> 2) << 4);    // 0,2,4,6,16,18,20,22 (+koff)
}

__device__ __forceinline__ v16bf load_frag_a(const bf16* __restrict__ base, int ld) {
  int lane = threadIdx.x & 31;
  int row  = lane & 15;
  v16bf f;
#pragma unroll
  for (int p = 0; p < 8; ++p) {
    int k = frag_k(lane, p);
    bf16x2 pr = *(const bf16x2*)(base + (size_t)row * ld + k);
    f[2 * p]     = pr.x;
    f[2 * p + 1] = pr.y;
  }
  return f;
}

__device__ __forceinline__ v8f wmma_bf16(v16bf a, v16bf b, v8f c) {
  return __builtin_amdgcn_wmma_f32_16x16x32_bf16(false, a, false, b, (short)0, c,
                                                 false, false);
}

__device__ __forceinline__ bf16 u16_as_bf16(unsigned short u) {
  union { unsigned short u; bf16 h; } c;
  c.u = u;
  return c.h;
}

// Low 32 bits of a generic LDS pointer == LDS byte offset (aperture is in the
// high dword), which is what the async-to-LDS VDST operand wants.
__device__ __forceinline__ uint32_t lds_off(const void* p) {
  return (uint32_t)(uintptr_t)p;
}

// GLOBAL_LOAD_ASYNC_TO_LDS_B128 (GVS mode): global data at saddr + vaddr goes to LDS at vdst,
// 16 bytes per lane, tracked by ASYNCcnt.
__device__ __forceinline__ void async_copy16(uint32_t lds_addr, const void* base,
                                             int byte_off) {
  asm volatile("global_load_async_to_lds_b128 %0, %1, %2"
               :: "v"(lds_addr), "v"(byte_off), "s"(base)
               : "memory");
}
__device__ __forceinline__ void wait_async0() {
  asm volatile("s_wait_asynccnt 0x0" ::: "memory");
}

// ---------------------------------------------------------------------------
// fp32 -> (hi, lo) bf16 split
// ---------------------------------------------------------------------------
__global__ void split_kernel(const float* __restrict__ src, bf16* __restrict__ hi,
                             bf16* __restrict__ lo, int n) {
  int i = blockIdx.x * blockDim.x + threadIdx.x;
  if (i < n) {
    float v = src[i];
    bf16 h = (bf16)v;
    hi[i] = h;
    lo[i] = (bf16)(v - (float)h);
  }
}

// ---------------------------------------------------------------------------
// C = A @ B + bias, bf16x3 split GEMM, double-buffered LDS.
// A tiles: async global->LDS DMA. B tiles: coalesced dword loads + transposed
// ds_store (stores issued after compute so load latency hides under WMMA).
// Workgroup tile 128x128, 8 waves (2x4), wave tile 64x32, K step 32.
// ---------------------------------------------------------------------------
#define LDSA 40  // 80B rows: 16B-aligned for b128 async, conflict-free frags
#define LDSB 34  // 68B rows for transposed B / P staging

__global__ __launch_bounds__(256) void gemm_bf16x3(
    const bf16* __restrict__ Ahi, const bf16* __restrict__ Alo,
    const bf16* __restrict__ Bhi, const bf16* __restrict__ Blo,
    const float* __restrict__ bias, float* __restrict__ Cf,
    bf16* __restrict__ Chi, bf16* __restrict__ Clo, int M, int N, int K) {
  __shared__ bf16 sAh[2][128][LDSA];
  __shared__ bf16 sAl[2][128][LDSA];
  __shared__ bf16 sBh[2][128][LDSB];  // transposed: [col][k]
  __shared__ bf16 sBl[2][128][LDSB];

  const int t    = threadIdx.x;
  const int wave = t >> 5;
  const int lane = t & 31;
  const int wr   = wave >> 2;  // 0..1
  const int wc   = wave & 3;   // 0..3
  const int row0 = blockIdx.y * 128;
  const int col0 = blockIdx.x * 128;

  // ---- staging helpers ----
  auto stageA = [&](int k0, int bsel) {  // 128 rows x 64B, 512 x 16B segments
    const bf16* bh = Ahi + (size_t)row0 * K + k0;
    const bf16* bl = Alo + (size_t)row0 * K + k0;
#pragma unroll
    for (int i = 0; i < 2; ++i) {
      int lin = t + i * 256;
      int r   = lin >> 2;
      int sg  = lin & 3;
      int go  = (r * K + sg * 8) * 2;
      async_copy16(lds_off(&sAh[bsel][r][sg * 8]), bh, go);
      async_copy16(lds_off(&sAl[bsel][r][sg * 8]), bl, go);
    }
  };
  auto loadB = [&](int k0, uint32_t* bh_, uint32_t* bl_) {
#pragma unroll
    for (int i = 0; i < 8; ++i) {
      int lin = t + i * 256;
      int kk  = lin >> 6;
      int dc  = lin & 63;
      size_t g = (size_t)(k0 + kk) * N + col0 + dc * 2;
      bh_[i] = *(const uint32_t*)(Bhi + g);
      bl_[i] = *(const uint32_t*)(Blo + g);
    }
  };
  auto storeB = [&](int bsel, const uint32_t* bh_, const uint32_t* bl_) {
#pragma unroll
    for (int i = 0; i < 8; ++i) {
      int lin = t + i * 256;
      int kk  = lin >> 6;
      int dc  = lin & 63;
      sBh[bsel][dc * 2][kk]     = u16_as_bf16((unsigned short)(bh_[i] & 0xffff));
      sBh[bsel][dc * 2 + 1][kk] = u16_as_bf16((unsigned short)(bh_[i] >> 16));
      sBl[bsel][dc * 2][kk]     = u16_as_bf16((unsigned short)(bl_[i] & 0xffff));
      sBl[bsel][dc * 2 + 1][kk] = u16_as_bf16((unsigned short)(bl_[i] >> 16));
    }
  };

  v8f acc[4][2] = {};

  // ---- prologue: tile 0 into buffer 0 ----
  {
    uint32_t bh_[8], bl_[8];
    stageA(0, 0);
    loadB(0, bh_, bl_);
    storeB(0, bh_, bl_);
    wait_async0();
    __syncthreads();
  }

  const int nk = K >> 5;
  for (int ti = 0; ti < nk; ++ti) {
    const int  bsel = ti & 1;
    const bool more = (ti + 1) < nk;
    uint32_t nbh[8], nbl[8];
    if (more) {
      stageA((ti + 1) * 32, bsel ^ 1);     // async DMA runs under the WMMAs
      loadB((ti + 1) * 32, nbh, nbl);      // issue loads; consume after compute
    }

    v16bf bfh[2], bfl[2];
#pragma unroll
    for (int c = 0; c < 2; ++c) {
      bfh[c] = load_frag_a(&sBh[bsel][wc * 32 + c * 16][0], LDSB);
      bfl[c] = load_frag_a(&sBl[bsel][wc * 32 + c * 16][0], LDSB);
    }
#pragma unroll
    for (int r = 0; r < 4; ++r) {
      v16bf afh = load_frag_a(&sAh[bsel][wr * 64 + r * 16][0], LDSA);
      v16bf afl = load_frag_a(&sAl[bsel][wr * 64 + r * 16][0], LDSA);
#pragma unroll
      for (int c = 0; c < 2; ++c) {
        acc[r][c] = wmma_bf16(afh, bfh[c], acc[r][c]);
        acc[r][c] = wmma_bf16(afh, bfl[c], acc[r][c]);
        acc[r][c] = wmma_bf16(afl, bfh[c], acc[r][c]);
      }
    }

    if (more) storeB(bsel ^ 1, nbh, nbl);
    wait_async0();
    __syncthreads();
  }

  const int colb = lane & 15;
  const int rowb = (lane >> 4) << 3;
#pragma unroll
  for (int r = 0; r < 4; ++r)
#pragma unroll
    for (int c = 0; c < 2; ++c)
#pragma unroll
      for (int v = 0; v < 8; ++v) {
        int row = row0 + wr * 64 + r * 16 + rowb + v;
        int col = col0 + wc * 32 + c * 16 + colb;
        float val = acc[r][c][v] + bias[col];
        size_t idx = (size_t)row * N + col;
        if (Cf) Cf[idx] = val;
        if (Chi) {
          bf16 h = (bf16)val;
          Chi[idx] = h;
          Clo[idx] = (bf16)(val - (float)h);
        }
      }
}

// ---------------------------------------------------------------------------
// Flash attention, non-causal. qkv split buffers: [B*L][6144] rows,
// q cols [0,2048), k cols [2048,4096), v cols [4096,6144); head h -> +h*128.
// One wave owns 16 query rows; 8 waves/WG share cooperatively staged K/V
// tiles (K: async global->LDS DMA, V: transposed ds_store). Grid (16, H, B).
// ---------------------------------------------------------------------------
__global__ __launch_bounds__(256) void attention_kernel(
    const bf16* __restrict__ qh_, const bf16* __restrict__ ql_,
    bf16* __restrict__ out_hi, bf16* __restrict__ out_lo) {
  const int L = 2048, E = 128, LD = 6144, D = 2048;
  const int t    = threadIdx.x;
  const int wave = t >> 5;
  const int lane = t & 31;
  const int b  = blockIdx.z;
  const int h  = blockIdx.y;
  const int q0 = blockIdx.x * 128 + wave * 16;

  __shared__ bf16 sKh[32][136], sKl[32][136];  // [key][e], 272B rows (16B mult)
  __shared__ bf16 sVh[128][LDSA], sVl[128][LDSA];  // transposed: [e][key]
  __shared__ bf16 sPh[8][16][LDSB], sPl[8][16][LDSB];

  // Q fragments, kept in registers for the whole key loop
  v16bf Qhf[4], Qlf[4];
#pragma unroll
  for (int kc = 0; kc < 4; ++kc) {
    size_t qb = (size_t)(b * L + q0) * LD + h * E + kc * 32;
    Qhf[kc] = load_frag_a(qh_ + qb, LD);
    Qlf[kc] = load_frag_a(ql_ + qb, LD);
  }

  v8f O[8] = {};
  float m_[8], l_[8];
#pragma unroll
  for (int v = 0; v < 8; ++v) { m_[v] = -INFINITY; l_[v] = 0.0f; }

  const float scale = 0.08838834764831845f;  // 1/sqrt(128)

  for (int kb = 0; kb < L; kb += 32) {
    // ----- cooperative staging of K (async DMA) and V^T (dword+scatter) ----
    {
      const bf16* kbh = qh_ + (size_t)(b * L + kb) * LD + 2048 + h * E;
      const bf16* kbl = ql_ + (size_t)(b * L + kb) * LD + 2048 + h * E;
#pragma unroll
      for (int i = 0; i < 2; ++i) {  // 32 rows x 256B = 512 x 16B segs
        int lin = t + i * 256;
        int r   = lin >> 4;
        int sg  = lin & 15;
        int go  = (r * LD + sg * 8) * 2;
        async_copy16(lds_off(&sKh[r][sg * 8]), kbh, go);
        async_copy16(lds_off(&sKl[r][sg * 8]), kbl, go);
      }
      uint32_t vh_[8], vl_[8];
#pragma unroll
      for (int i = 0; i < 8; ++i) {  // 32 rows x 64 dwords
        int lin = t + i * 256;
        int kk  = lin >> 6;
        int dc  = lin & 63;
        size_t g = (size_t)(b * L + kb + kk) * LD + 4096 + h * E + dc * 2;
        vh_[i] = *(const uint32_t*)(qh_ + g);
        vl_[i] = *(const uint32_t*)(ql_ + g);
      }
#pragma unroll
      for (int i = 0; i < 8; ++i) {
        int lin = t + i * 256;
        int kk  = lin >> 6;
        int dc  = lin & 63;
        sVh[dc * 2][kk]     = u16_as_bf16((unsigned short)(vh_[i] & 0xffff));
        sVh[dc * 2 + 1][kk] = u16_as_bf16((unsigned short)(vh_[i] >> 16));
        sVl[dc * 2][kk]     = u16_as_bf16((unsigned short)(vl_[i] & 0xffff));
        sVl[dc * 2 + 1][kk] = u16_as_bf16((unsigned short)(vl_[i] >> 16));
      }
      wait_async0();
      __syncthreads();
    }

    // ----- S = Q @ K^T (keys kb..kb+15 and kb+16..kb+31) -------------------
    v8f S0 = {}, S1 = {};
#pragma unroll
    for (int kc = 0; kc < 4; ++kc) {
      v16bf K0h = load_frag_a(&sKh[0][kc * 32], 136);
      v16bf K0l = load_frag_a(&sKl[0][kc * 32], 136);
      v16bf K1h = load_frag_a(&sKh[16][kc * 32], 136);
      v16bf K1l = load_frag_a(&sKl[16][kc * 32], 136);
      S0 = wmma_bf16(Qhf[kc], K0h, S0);
      S0 = wmma_bf16(Qhf[kc], K0l, S0);
      S0 = wmma_bf16(Qlf[kc], K0h, S0);
      S1 = wmma_bf16(Qhf[kc], K1h, S1);
      S1 = wmma_bf16(Qhf[kc], K1l, S1);
      S1 = wmma_bf16(Qlf[kc], K1h, S1);
    }

    // ----- online softmax update (row stats across the 16 lanes of a half) -
    float alpha[8];
#pragma unroll
    for (int v = 0; v < 8; ++v) {
      float s0 = S0[v] * scale, s1 = S1[v] * scale;
      float mx = fmaxf(s0, s1);
      mx = fmaxf(mx, __shfl_xor(mx, 1, 32));
      mx = fmaxf(mx, __shfl_xor(mx, 2, 32));
      mx = fmaxf(mx, __shfl_xor(mx, 4, 32));
      mx = fmaxf(mx, __shfl_xor(mx, 8, 32));
      float mnew = fmaxf(m_[v], mx);
      float p0 = __expf(s0 - mnew);
      float p1 = __expf(s1 - mnew);
      S0[v] = p0;
      S1[v] = p1;
      float sum = p0 + p1;
      sum += __shfl_xor(sum, 1, 32);
      sum += __shfl_xor(sum, 2, 32);
      sum += __shfl_xor(sum, 4, 32);
      sum += __shfl_xor(sum, 8, 32);
      alpha[v] = __expf(m_[v] - mnew);
      l_[v] = l_[v] * alpha[v] + sum;
      m_[v] = mnew;
    }
#pragma unroll
    for (int ec = 0; ec < 8; ++ec)
#pragma unroll
      for (int v = 0; v < 8; ++v) O[ec][v] *= alpha[v];

    // ----- stage P (hi/lo) through wave-private LDS: C -> A layout ---------
    const int prow = (lane >> 4) << 3;
    const int pcol = lane & 15;
#pragma unroll
    for (int v = 0; v < 8; ++v) {
      float p0 = S0[v], p1 = S1[v];
      bf16 h0 = (bf16)p0, h1 = (bf16)p1;
      sPh[wave][prow + v][pcol]      = h0;
      sPh[wave][prow + v][16 + pcol] = h1;
      sPl[wave][prow + v][pcol]      = (bf16)(p0 - (float)h0);
      sPl[wave][prow + v][16 + pcol] = (bf16)(p1 - (float)h1);
    }
    v16bf Pa_h = load_frag_a(&sPh[wave][0][0], LDSB);
    v16bf Pa_l = load_frag_a(&sPl[wave][0][0], LDSB);

    // ----- O += P @ V (V^T staged in LDS) ----------------------------------
#pragma unroll
    for (int ec = 0; ec < 8; ++ec) {
      v16bf Vh = load_frag_a(&sVh[ec * 16][0], LDSA);
      v16bf Vl = load_frag_a(&sVl[ec * 16][0], LDSA);
      O[ec] = wmma_bf16(Pa_h, Vh, O[ec]);
      O[ec] = wmma_bf16(Pa_h, Vl, O[ec]);
      O[ec] = wmma_bf16(Pa_l, Vh, O[ec]);
    }
    __syncthreads();  // everyone done reading K/V before next overwrite
  }

  // ----- normalize and emit split output [B*L][2048] ------------------------
  const int rowb = (lane >> 4) << 3;
  const int colb = lane & 15;
#pragma unroll
  for (int v = 0; v < 8; ++v) {
    float inv = 1.0f / l_[v];
    int row = q0 + rowb + v;
#pragma unroll
    for (int ec = 0; ec < 8; ++ec) {
      float val = O[ec][v] * inv;
      size_t idx = (size_t)(b * L + row) * D + h * E + ec * 16 + colb;
      bf16 hh = (bf16)val;
      out_hi[idx] = hh;
      out_lo[idx] = (bf16)(val - (float)hh);
    }
  }
}

// ---------------------------------------------------------------------------
// Host side
// ---------------------------------------------------------------------------
extern "C" void kernel_launch(void* const* d_in, const int* in_sizes, int n_in,
                              void* d_out, int out_size, void* d_ws, size_t ws_size,
                              hipStream_t stream) {
  const float* x     = (const float*)d_in[0];
  const float* w_qkv = (const float*)d_in[1];
  const float* b_qkv = (const float*)d_in[2];
  const float* w_out = (const float*)d_in[3];
  const float* b_out = (const float*)d_in[4];
  float* out = (float*)d_out;

  const int BL = 2 * 2048;  // 4096 tokens
  const int D = 2048, TD = 6144;

  char* p = (char*)d_ws;
  auto take = [&](size_t elems) {
    bf16* r = (bf16*)p;
    p += (elems * sizeof(bf16) + 255) & ~(size_t)255;
    return r;
  };
  bf16* xh  = take((size_t)BL * D);
  bf16* xl  = take((size_t)BL * D);
  bf16* wqh = take((size_t)D * TD);
  bf16* wql = take((size_t)D * TD);
  bf16* woh = take((size_t)D * D);
  bf16* wol = take((size_t)D * D);
  bf16* qvh = take((size_t)BL * TD);
  bf16* qvl = take((size_t)BL * TD);
  bf16* ah  = take((size_t)BL * D);
  bf16* al  = take((size_t)BL * D);

  auto split = [&](const float* s, bf16* hi, bf16* lo, size_t n) {
    split_kernel<<<dim3((unsigned)((n + 511) / 512)), dim3(512), 0, stream>>>(
        s, hi, lo, (int)n);
  };
  split(x, xh, xl, (size_t)BL * D);
  split(w_qkv, wqh, wql, (size_t)D * TD);
  split(w_out, woh, wol, (size_t)D * D);

  // QKV projection: [4096,2048] @ [2048,6144] + b_qkv -> split bf16 qkv
  gemm_bf16x3<<<dim3(TD / 128, BL / 128), dim3(256), 0, stream>>>(
      xh, xl, wqh, wql, b_qkv, nullptr, qvh, qvl, BL, TD, D);

  // attention -> split bf16 [4096,2048]
  attention_kernel<<<dim3(2048 / 128, 16, 2), dim3(256), 0, stream>>>(qvh, qvl, ah, al);

  // output projection: [4096,2048] @ [2048,2048] + b_out -> fp32 d_out
  gemm_bf16x3<<<dim3(D / 128, BL / 128), dim3(256), 0, stream>>>(
      ah, al, woh, wol, b_out, out, nullptr, nullptr, BL, D, D);
}
